// VectorQuantizer_68427418960144
// MI455X (gfx1250) — compile-verified, matrix operand order checked by reading, not run on hardware
//
#include <hip/hip_runtime.h>
#include <math.h>

// ---------------------------------------------------------------------------
// VQ-VAE vector quantizer for MI455X (gfx1250, wave32, WMMA).
// N=65536 z-vectors, K=4096 codes, D=64.
// Score GEMM via V_WMMA_F32_16X16X32_BF16 with bf16x3 split precision
// (hi*hi + hi*lo + lo*hi) for fp32-quality argmin at ~bf16/3 throughput.
// Each wave owns 32 rows (2 A tiles) so 8 ds_load_b128 feed 12 WMMAs.
// B macro-tiles (64 codes) double-buffered in LDS via async loads-to-LDS
// (GLOBAL_LOAD_ASYNC_TO_LDS_B128, tracked by ASYNCcnt).
// ---------------------------------------------------------------------------

typedef __attribute__((ext_vector_type(16))) __bf16 v16bf;
typedef __attribute__((ext_vector_type(8)))  float  v8f;
typedef int vqi4 __attribute__((__vector_size__(16)));   // matches builtin pointee

#if __has_builtin(__builtin_amdgcn_global_load_async_to_lds_b128) && \
    __has_builtin(__builtin_amdgcn_s_wait_asynccnt)
#define VQ_ASYNC_LDS 1
#else
#define VQ_ASYNC_LDS 0
#endif

union BF16x16 {
    v16bf          v;
    unsigned short u[16];
    uint4          q[2];
};

__device__ __forceinline__ unsigned short f2bf(float x) {
    unsigned int b = __float_as_uint(x);
    unsigned int r = b + 0x7FFFu + ((b >> 16) & 1u);   // round-to-nearest-even
    return (unsigned short)(r >> 16);
}
__device__ __forceinline__ float bf2f(unsigned short h) {
    return __uint_as_float(((unsigned int)h) << 16);
}

__device__ __forceinline__ v8f wmma_bf16(const BF16x16& a, const BF16x16& b, v8f c) {
    // 8 args: (neg_a, A, neg_b, B, c_mod, C, reuse_a, reuse_b)
    return __builtin_amdgcn_wmma_f32_16x16x32_bf16(false, a.v, false, b.v,
                                                   (short)0, c, false, false);
}

#if VQ_ASYNC_LDS
__device__ __forceinline__ void async_copy_b128(const void* g, void* l) {
    __builtin_amdgcn_global_load_async_to_lds_b128(
        (__attribute__((address_space(1))) vqi4*)(g),
        (__attribute__((address_space(3))) vqi4*)(l),
        0, 0);
}
#endif

// ---------------------------------------------------------------------------
// Kernel 0: zero counts[K] and the SSE accumulator.
// ---------------------------------------------------------------------------
__global__ void vq_zero(float* counts, float* sacc, int K) {
    int i = blockIdx.x * blockDim.x + threadIdx.x;
    if (i < K) counts[i] = 0.0f;
    if (i == 0) *sacc = 0.0f;
}

// ---------------------------------------------------------------------------
// Kernel 1: embedding prep. One wave per code: split e -> bf16 hi/lo and
// compute e2[k] = sum_d e[k,d]^2 (fp32, shfl reduce).
// ---------------------------------------------------------------------------
__global__ __launch_bounds__(256) void vq_prep(const float* __restrict__ emb,
                                               unsigned short* __restrict__ ehi,
                                               unsigned short* __restrict__ elo,
                                               float* __restrict__ e2, int K) {
    int w    = threadIdx.x >> 5;
    int lane = threadIdx.x & 31;
    int k    = blockIdx.x * 8 + w;
    if (k >= K) return;
    float s = 0.0f;
#pragma unroll
    for (int j = 0; j < 2; j++) {
        int d = lane + j * 32;
        float x = emb[k * 64 + d];
        unsigned short h = f2bf(x);
        ehi[k * 64 + d] = h;
        elo[k * 64 + d] = f2bf(x - bf2f(h));
        s += x * x;
    }
#pragma unroll
    for (int m = 16; m >= 1; m >>= 1) s += __shfl_xor(s, m, 32);
    if (lane == 0) e2[k] = s;
}

// ---------------------------------------------------------------------------
// Kernel 2: main. Workgroup = 256 threads = 8 waves; each wave owns 32 rows of
// z (two 16-row A tiles resident in VGPRs as bf16 hi/lo, WMMA A layout).
// The workgroup sweeps K in macro-chunks of 64 codes, double-buffered in LDS
// via async load-to-LDS; each wave runs 4 sub-chunks of 16 codes: 12 WMMAs
// per sub-chunk (bf16x3 x 2 row tiles), then per-lane running argmin of
// dist = e2 - 2*score. Final butterfly reduce, gather z_q, SSE, counts.
// ---------------------------------------------------------------------------
__global__ __launch_bounds__(256) void vq_main(const float* __restrict__ z,
                                               const float* __restrict__ emb,
                                               const unsigned short* __restrict__ ehi,
                                               const unsigned short* __restrict__ elo,
                                               const float* __restrict__ e2,
                                               float* __restrict__ counts,
                                               float* __restrict__ sacc,
                                               float* __restrict__ out_zq,
                                               float* __restrict__ out_idx,
                                               int K) {
    __shared__ __align__(16) float          zt[256 * 64];          // 64 KB z tile
    __shared__ __align__(16) unsigned short bh[2][64 * 64];        // 2 x 8 KB B hi
    __shared__ __align__(16) unsigned short bl[2][64 * 64];        // 2 x 8 KB B lo

    const int tid   = threadIdx.x;
    const int lane  = tid & 31;
    const int w     = tid >> 5;
    const int wgRow = blockIdx.x * 256;

    // Stage this workgroup's 256x64 fp32 z tile into LDS.
    {
        const uint4* zg = (const uint4*)(z + (size_t)wgRow * 64);
        uint4* zs = (uint4*)zt;
        for (int i = tid; i < 256 * 64 / 4; i += 256) zs[i] = zg[i];
    }

    // Issue B stage for macro-tile 0 into buffer 0 (16 KB: 4 x b128 per thread).
    {
        const uint4* gh = (const uint4*)ehi;
        const uint4* gl = (const uint4*)elo;
        uint4* sh = (uint4*)bh[0];
        uint4* sl = (uint4*)bl[0];
#if VQ_ASYNC_LDS
#pragma unroll
        for (int i = 0; i < 2; i++) {
            async_copy_b128(gh + tid + i * 256, sh + tid + i * 256);
            async_copy_b128(gl + tid + i * 256, sl + tid + i * 256);
        }
#else
        for (int i = tid; i < 512; i += 256) { sh[i] = gh[i]; sl[i] = gl[i]; }
#endif
    }
#if VQ_ASYNC_LDS
    __builtin_amdgcn_s_wait_asynccnt(0);
#endif
    __syncthreads();

    // Build resident A fragments (2 tiles x 16 rows x 64 dims, bf16 hi/lo) in
    // WMMA A layout: lane l<16 -> row l, K-offsets {0..7,16..23}+32*kc; l>=16 -> +8.
    const int kofs = (lane < 16) ? 0 : 8;
    BF16x16 ah[2][2], al[2][2];                  // [row tile][kc]
#pragma unroll
    for (int t = 0; t < 2; t++) {
        const int arow = w * 32 + t * 16 + (lane & 15);
#pragma unroll
        for (int kc = 0; kc < 2; kc++) {
            const float* zr = &zt[arow * 64 + kc * 32 + kofs];
#pragma unroll
            for (int e = 0; e < 8; e++) {
                float x0 = zr[e];
                float x1 = zr[16 + e];
                unsigned short h0 = f2bf(x0), h1 = f2bf(x1);
                ah[t][kc].u[e]     = h0;
                ah[t][kc].u[8 + e] = h1;
                al[t][kc].u[e]     = f2bf(x0 - bf2f(h0));
                al[t][kc].u[8 + e] = f2bf(x1 - bf2f(h1));
            }
        }
    }

    float minv[2][8];
    int   mini[2][8];
#pragma unroll
    for (int t = 0; t < 2; t++)
#pragma unroll
        for (int i = 0; i < 8; i++) { minv[t][i] = 3.4e38f; mini[t][i] = 0; }

    const int nMacro = K / 64;
    for (int mc = 0; mc < nMacro; mc++) {
        const int buf = mc & 1;
        // Issue async stage of next macro-tile into the other buffer.
        if (mc + 1 < nMacro) {
            const uint4* gh = (const uint4*)(ehi + (size_t)(mc + 1) * 64 * 64);
            const uint4* gl = (const uint4*)(elo + (size_t)(mc + 1) * 64 * 64);
            uint4* sh = (uint4*)bh[buf ^ 1];
            uint4* sl = (uint4*)bl[buf ^ 1];
#if VQ_ASYNC_LDS
#pragma unroll
            for (int i = 0; i < 2; i++) {
                async_copy_b128(gh + tid + i * 256, sh + tid + i * 256);
                async_copy_b128(gl + tid + i * 256, sl + tid + i * 256);
            }
#else
            for (int i = tid; i < 512; i += 256) { sh[i] = gh[i]; sl[i] = gl[i]; }
#endif
        }

        const unsigned short* bhp = bh[buf];
        const unsigned short* blp = bl[buf];
#pragma unroll
        for (int sc = 0; sc < 4; sc++) {
            const int ccol = sc * 16 + (lane & 15);   // LDS-local code
            const int gcol = mc * 64 + ccol;          // global code index
            BF16x16 bhv[2], blv[2];
#pragma unroll
            for (int kc = 0; kc < 2; kc++) {
                const int ks = kc * 32 + kofs;
                bhv[kc].q[0] = *(const uint4*)&bhp[ccol * 64 + ks];
                bhv[kc].q[1] = *(const uint4*)&bhp[ccol * 64 + ks + 16];
                blv[kc].q[0] = *(const uint4*)&blp[ccol * 64 + ks];
                blv[kc].q[1] = *(const uint4*)&blp[ccol * 64 + ks + 16];
            }
            v8f c0 = {}, c1 = {};
            c0 = wmma_bf16(ah[0][0], bhv[0], c0);   // hi * hi
            c1 = wmma_bf16(ah[1][0], bhv[0], c1);
            c0 = wmma_bf16(ah[0][1], bhv[1], c0);
            c1 = wmma_bf16(ah[1][1], bhv[1], c1);
            c0 = wmma_bf16(al[0][0], bhv[0], c0);   // lo * hi
            c1 = wmma_bf16(al[1][0], bhv[0], c1);
            c0 = wmma_bf16(al[0][1], bhv[1], c0);
            c1 = wmma_bf16(al[1][1], bhv[1], c1);
            c0 = wmma_bf16(ah[0][0], blv[0], c0);   // hi * lo
            c1 = wmma_bf16(ah[1][0], blv[0], c1);
            c0 = wmma_bf16(ah[0][1], blv[1], c0);
            c1 = wmma_bf16(ah[1][1], blv[1], c1);

            const float ev = e2[gcol];
#pragma unroll
            for (int i = 0; i < 8; i++) {
                float d0 = ev - 2.0f * c0[i];
                if (d0 < minv[0][i]) { minv[0][i] = d0; mini[0][i] = gcol; }
                float d1 = ev - 2.0f * c1[i];
                if (d1 < minv[1][i]) { minv[1][i] = d1; mini[1][i] = gcol; }
            }
        }
#if VQ_ASYNC_LDS
        __builtin_amdgcn_s_wait_asynccnt(0);
#endif
        __syncthreads();
    }

    // Butterfly min-reduce within each 16-lane half (halves = row groups 0-7 / 8-15).
#pragma unroll
    for (int m = 8; m >= 1; m >>= 1) {
#pragma unroll
        for (int t = 0; t < 2; t++)
#pragma unroll
            for (int i = 0; i < 8; i++) {
                float ov = __shfl_xor(minv[t][i], m, 32);
                int   oi = __shfl_xor(mini[t][i], m, 32);
                if (ov < minv[t][i] || (ov == minv[t][i] && oi < mini[t][i])) {
                    minv[t][i] = ov; mini[t][i] = oi;
                }
            }
    }

    // Gather z_q = embedding[idx] (fp32), write z_q_st (== z_q), SSE, counts, indices.
    const int half = lane >> 4;   // 0 -> rows i, 1 -> rows i+8
    const int q    = lane & 15;   // 16 lanes x float4 cover D=64
    float sse = 0.0f;
#pragma unroll
    for (int t = 0; t < 2; t++) {
#pragma unroll
        for (int i = 0; i < 8; i++) {
            const int rloc = w * 32 + t * 16 + half * 8 + i;
            const int grow = wgRow + rloc;
            const int idx  = mini[t][i];
            const float4 e4 = *(const float4*)(emb + (size_t)idx * 64 + q * 4);
            const float4 z4 = *(const float4*)&zt[rloc * 64 + q * 4];
            *(float4*)(out_zq + (size_t)grow * 64 + q * 4) = e4;
            float dx = e4.x - z4.x, dy = e4.y - z4.y, dz = e4.z - z4.z, dw = e4.w - z4.w;
            sse += dx * dx + dy * dy + dz * dz + dw * dw;
            if (q == 0) {
                out_idx[grow] = (float)idx;
                atomicAdd(&counts[idx], 1.0f);
            }
        }
    }
#pragma unroll
    for (int m = 16; m >= 1; m >>= 1) sse += __shfl_xor(sse, m, 32);
    if (lane == 0) atomicAdd(sacc, sse);
}

// ---------------------------------------------------------------------------
// Kernel 3: finalize scalars. losses from SSE; perplexity/active from counts.
// ---------------------------------------------------------------------------
__global__ __launch_bounds__(256) void vq_finalize(const float* __restrict__ counts,
                                                   const float* __restrict__ sacc,
                                                   float* __restrict__ out,
                                                   int N, int K) {
    __shared__ float se[256], sa[256];
    const int tid = threadIdx.x;
    const float invN = 1.0f / (float)N;
    float ent = 0.0f, act = 0.0f;
    for (int k = tid; k < K; k += 256) {
        float p = counts[k] * invN;
        ent += p * logf(p + 1e-10f);
        if (p > 0.0f) act += 1.0f;
    }
    se[tid] = ent; sa[tid] = act;
    __syncthreads();
    for (int s = 128; s > 0; s >>= 1) {
        if (tid < s) { se[tid] += se[tid + s]; sa[tid] += sa[tid + s]; }
        __syncthreads();
    }
    if (tid == 0) {
        const size_t base = (size_t)N * 64;
        float cb = (*sacc) / ((float)N * 64.0f);   // codebook_loss == commit_loss
        out[base + 0] = cb + 0.25f * cb;           // vq_loss
        out[base + 1] = cb;                        // codebook_loss
        out[base + 2] = cb;                        // commit_loss
        out[base + 3 + (size_t)N]     = expf(-se[0]);  // perplexity
        out[base + 3 + (size_t)N + 1] = sa[0];         // active_codes
    }
}

// ---------------------------------------------------------------------------
extern "C" void kernel_launch(void* const* d_in, const int* in_sizes, int n_in,
                              void* d_out, int out_size, void* d_ws, size_t ws_size,
                              hipStream_t stream) {
    const float* z   = (const float*)d_in[0];
    const float* emb = (const float*)d_in[1];
    const int N = in_sizes[0] / 64;     // 65536
    const int K = in_sizes[1] / 64;     // 4096

    // Workspace layout: ehi[K*64] u16 | elo[K*64] u16 | e2[K] f32 | counts[K] f32 | sacc f32
    unsigned short* ehi = (unsigned short*)d_ws;
    unsigned short* elo = ehi + (size_t)K * 64;
    float* e2     = (float*)(elo + (size_t)K * 64);
    float* counts = e2 + K;
    float* sacc   = counts + K;

    float* out     = (float*)d_out;
    float* out_zq  = out;
    float* out_idx = out + (size_t)N * 64 + 3;

    hipLaunchKernelGGL(vq_zero, dim3((K + 1 + 255) / 256), dim3(256), 0, stream,
                       counts, sacc, K);
    hipLaunchKernelGGL(vq_prep, dim3(K / 8), dim3(256), 0, stream,
                       emb, ehi, elo, e2, K);
    hipLaunchKernelGGL(vq_main, dim3(N / 256), dim3(256), 0, stream,
                       z, emb, ehi, elo, e2, counts, sacc, out_zq, out_idx, K);
    hipLaunchKernelGGL(vq_finalize, dim3(1), dim3(256), 0, stream,
                       counts, sacc, out, N, K);
}
